// MultiTurnInferenceDecoder_Z_84482006712453
// MI455X (gfx1250) — matrix operand order repaired and checked by reading, not compiled, not run on hardware
//
#include <hip/hip_runtime.h>

// ---------------- problem sizes ----------------
#define E_DIM 512
#define H_DIM 1024
#define V_DIM 32000
#define B_DIM 64
#define TU_DIM 128
#define TM_DIM 128
#define TZ_DIM 64
#define EPSF 1e-10f

// ---------------- vector types ----------------
typedef __attribute__((ext_vector_type(16))) __bf16         v16bf;
typedef __attribute__((ext_vector_type(8)))  float          v8f;
typedef __attribute__((ext_vector_type(8)))  unsigned short u16x8;
typedef __attribute__((ext_vector_type(4)))  unsigned int   u32x4;
typedef __attribute__((ext_vector_type(8)))  int            i32x8;
typedef __attribute__((ext_vector_type(4)))  int            i32x4;

__device__ __forceinline__ unsigned short f2bfu(float f) {
  unsigned u = __builtin_bit_cast(unsigned, f);
  return (unsigned short)((u + 0x7FFFu + ((u >> 16) & 1u)) >> 16);  // RNE
}
__device__ __forceinline__ __bf16 f2bf(float f) {
  return __builtin_bit_cast(__bf16, f2bfu(f));
}
__device__ __forceinline__ __bf16 bfbits(unsigned short h) {
  return __builtin_bit_cast(__bf16, h);
}

// =====================================================================
// Kernel 1: single-step GRU (small: 300 M MAC, 19 MB weights) — VALU.
// Writes h1 to g (fp32), g_bf (bf16), and both gru_out / new_hidden.
// =====================================================================
__device__ __forceinline__ float dot4(const float* a, const float* b, int n) {
  float s = 0.f;
  for (int i = 0; i < n; i += 4) {
    float4 x = *(const float4*)(a + i);
    float4 y = *(const float4*)(b + i);
    s = fmaf(x.x, y.x, s); s = fmaf(x.y, y.y, s);
    s = fmaf(x.z, y.z, s); s = fmaf(x.w, y.w, s);
  }
  return s;
}

__global__ __launch_bounds__(256) void gru_kernel(
    const float* __restrict__ x,   // (B,E)  embed_z[0]
    const float* __restrict__ h0,  // (B,H)  last_hidden[0]
    const float* __restrict__ W_ih, const float* __restrict__ W_hh,
    const float* __restrict__ b_ih, const float* __restrict__ b_hh,
    float* __restrict__ g, unsigned short* __restrict__ g_bf,
    float* __restrict__ gru_out, float* __restrict__ new_hidden) {
  int idx = blockIdx.x * blockDim.x + threadIdx.x;
  if (idx >= B_DIM * H_DIM) return;
  int b = idx / H_DIM, h = idx % H_DIM;
  const float* xb = x  + (size_t)b * E_DIM;
  const float* hb = h0 + (size_t)b * H_DIM;
  float gi[3], gh[3];
#pragma unroll
  for (int gt = 0; gt < 3; ++gt) {
    int row = gt * H_DIM + h;
    gi[gt] = dot4(xb, W_ih + (size_t)row * E_DIM, E_DIM) + b_ih[row];
    gh[gt] = dot4(hb, W_hh + (size_t)row * H_DIM, H_DIM) + b_hh[row];
  }
  float r  = 1.f / (1.f + __expf(-(gi[0] + gh[0])));
  float z  = 1.f / (1.f + __expf(-(gi[1] + gh[1])));
  float nn = tanhf(gi[2] + r * gh[2]);
  float h1 = (1.f - z) * nn + z * hb[h];
  g[idx] = h1; g_bf[idx] = f2bfu(h1);
  gru_out[idx] = h1; new_hidden[idx] = h1;
}

// =====================================================================
// WMMA fragment helpers (per CDNA5 ISA §7.12.2 layouts)
//   A 16x32 bf16:   lanes 0-15: elems 0..7 = K0..7,  8..15 = K16..23
//                   lanes16-31: elems 0..7 = K8..15, 8..15 = K24..31
//   B 32x16 bf16:   lane n = lane&15; lanes0-15 K=0..7,16..23, lanes16-31 K=8..15,24..31 (as A)
//   C/D 16x16 f32:  elem r -> row r + (lane>=16 ? 8:0), col = lane&15
// =====================================================================
__device__ __forceinline__ v16bf load_a_bf16(const unsigned short* base,
                                             size_t stride, int m, int kb) {
  const unsigned short* p = base + (size_t)m * stride + kb;
  u16x8 lo = *(const u16x8*)(p);
  u16x8 hi = *(const u16x8*)(p + 16);
  v16bf a;
#pragma unroll
  for (int i = 0; i < 8; ++i) { a[i] = bfbits(lo[i]); a[8 + i] = bfbits(hi[i]); }
  return a;
}
__device__ __forceinline__ v16bf load_a_f32lds(const float* row, int kb) {
  v16bf a;
#pragma unroll
  for (int i = 0; i < 8; ++i) { a[i] = f2bf(row[kb + i]); a[8 + i] = f2bf(row[kb + 16 + i]); }
  return a;
}
__device__ __forceinline__ v16bf load_b_f32(const float* p /*K p[0..7], p[16..23]*/) {
  v16bf bf;
#pragma unroll
  for (int i = 0; i < 16; ++i) bf[i] = f2bf(p[i < 8 ? i : i + 8]);
  return bf;
}

// =====================================================================
// Kernel 2: gen_score = g @ w1^T + b   (64 x 32000, K=1024), WMMA bf16.
// grid = V/128 blocks of 256 threads; each wave owns one 16-col N-tile,
// loops 4 M-tiles covering all 64 rows.
// =====================================================================
__global__ __launch_bounds__(256) void gen_score_kernel(
    const unsigned short* __restrict__ g_bf,  // (B,H) bf16
    const float* __restrict__ w1_W,           // (V,H)
    const float* __restrict__ w1_b,
    float* __restrict__ gen) {                // (B,V)
  int lane = threadIdx.x & 31, wid = threadIdx.x >> 5;
  int n0 = (blockIdx.x * 8 + wid) * 16;
  int n  = n0 + (lane & 15);
  v8f acc[4] = {};
  for (int kk = 0; kk < H_DIM; kk += 32) {
    const float* brow = w1_W + (size_t)n * H_DIM + kk + ((lane & 16) ? 8 : 0);
    __builtin_prefetch(brow + 32, 0, 1);
    v16bf bf = load_b_f32(brow);
    int kb = kk + ((lane & 16) ? 8 : 0);
#pragma unroll
    for (int mt = 0; mt < 4; ++mt) {
      v16bf a = load_a_bf16(g_bf, H_DIM, mt * 16 + (lane & 15), kb);
      acc[mt] = __builtin_amdgcn_wmma_f32_16x16x32_bf16(
          false, a, false, bf, (short)0, acc[mt], false, false);
    }
  }
  float bias = w1_b[n];
#pragma unroll
  for (int mt = 0; mt < 4; ++mt)
#pragma unroll
    for (int r = 0; r < 8; ++r) {
      int row = mt * 16 + r + ((lane & 16) ? 8 : 0);
      gen[(size_t)row * V_DIM + n] = acc[mt][r] + bias;
    }
}

// =====================================================================
// Kernel 3: fused copy-score  s[b,t] = sum_n tanh(enc[t,b,:]·Pw[n,:]+Pb[n])·g[b,n]
// One block per 16 rows of (T*B, H); rows share t, b = b0..b0+15.
// A row-block (16x1024 fp32, 64KB) staged to LDS by the Tensor Data Mover,
// then 8 waves × 8 N-tiles × 32 K-step WMMAs, epilogue fuses tanh + dot(g).
// =====================================================================
__global__ __launch_bounds__(256) void copy_score_kernel(
    const float* __restrict__ enc,  // (T,B,H)
    const float* __restrict__ Pw,   // (H,H)
    const float* __restrict__ Pb,   // (H)
    const float* __restrict__ g,    // (B,H)
    float* __restrict__ s_out,      // (B,T)
    int T) {
  __shared__ float atile[16 * H_DIM];  // 64 KB of 320 KB/WGP
  __shared__ float sacc[16];
  int lane = threadIdx.x & 31, wid = threadIdx.x >> 5;
  int r0 = blockIdx.x * 16;
  int b0 = r0 & (B_DIM - 1);
  int t0 = r0 / B_DIM;
  if (threadIdx.x < 16) sacc[threadIdx.x] = 0.f;

  if (threadIdx.x < 32) {
    // --- Tensor DMA descriptor (CDNA5 ISA) ---
    unsigned long long ga = (unsigned long long)(const void*)(enc + (size_t)r0 * H_DIM);
    unsigned lds = (unsigned)(unsigned long long)(void*)atile;
    u32x4 g0;
    g0[0] = 1u;                                            // count=1, user mode
    g0[1] = lds;                                           // lds_addr
    g0[2] = (unsigned)(ga & 0xFFFFFFFFu);                  // global_addr[31:0]
    g0[3] = (unsigned)((ga >> 32) & 0x1FFFFFFu) | (2u << 30);  // [56:32] | type=2
    unsigned td0 = H_DIM, td1 = (unsigned)(T * B_DIM);
    i32x8 g1;
    g1[0] = (int)(2u << 16);                               // data_size=4B
    g1[1] = (int)((td0 & 0xFFFFu) << 16);                  // tensor_dim0 lo
    g1[2] = (int)(((td0 >> 16) & 0xFFFFu) | ((td1 & 0xFFFFu) << 16));
    g1[3] = (int)(((td1 >> 16) & 0xFFFFu) | ((unsigned)H_DIM << 16)); // tile_dim0=1024
    g1[4] = (int)16u;                                      // tile_dim1=16, tile_dim2=0
    g1[5] = (int)(unsigned)H_DIM;                          // tensor_dim0_stride lo32
    g1[6] = 0;                                             // stride hi / dim1_stride lo
    g1[7] = 0;
    i32x4 z4 = {0, 0, 0, 0};
    i32x8 z8 = {0, 0, 0, 0, 0, 0, 0, 0};
    __builtin_amdgcn_tensor_load_to_lds(g0, g1, z4, z4, z8, 0);
    __builtin_amdgcn_s_wait_tensorcnt(0);
  }
  __syncthreads();

  float dot[8] = {0.f, 0.f, 0.f, 0.f, 0.f, 0.f, 0.f, 0.f};
  for (int ntl = 0; ntl < 8; ++ntl) {
    int nt = wid * 8 + ntl;
    int n  = nt * 16 + (lane & 15);
    v8f acc = {0.f, 0.f, 0.f, 0.f, 0.f, 0.f, 0.f, 0.f};
    for (int kk = 0; kk < H_DIM; kk += 32) {
      const float* brow = Pw + (size_t)n * H_DIM + kk + ((lane & 16) ? 8 : 0);
      v16bf bf = load_b_f32(brow);
      v16bf a  = load_a_f32lds(atile + (size_t)(lane & 15) * H_DIM,
                               kk + ((lane & 16) ? 8 : 0));
      acc = __builtin_amdgcn_wmma_f32_16x16x32_bf16(
          false, a, false, bf, (short)0, acc, false, false);
    }
    float pb = Pb[n];
#pragma unroll
    for (int r = 0; r < 8; ++r) {
      int mloc = r + ((lane & 16) ? 8 : 0);
      float c = tanhf(acc[r] + pb);
      dot[r] = fmaf(c, g[(size_t)(b0 + mloc) * H_DIM + n], dot[r]);
    }
  }
  // reduce across the 16 lanes of each half-wave
#pragma unroll
  for (int r = 0; r < 8; ++r) {
    float v = dot[r];
    v += __shfl_xor(v, 1, 32);
    v += __shfl_xor(v, 2, 32);
    v += __shfl_xor(v, 4, 32);
    v += __shfl_xor(v, 8, 32);
    dot[r] = v;
  }
  if ((lane & 15) == 0) {
    int base = (lane & 16) ? 8 : 0;
#pragma unroll
    for (int r = 0; r < 8; ++r) atomicAdd(&sacc[base + r], dot[r]);
  }
  __syncthreads();
  if (threadIdx.x < 16)
    s_out[(size_t)(b0 + threadIdx.x) * T + t0] = sacc[threadIdx.x];
}

// =====================================================================
// Kernel 4: per-row max-shifted exp (u/m branches): e = exp(s - max), in place.
// =====================================================================
__global__ __launch_bounds__(128) void softexp_kernel(float* __restrict__ s,
                                                      float* __restrict__ smax,
                                                      float* __restrict__ esum, int T) {
  __shared__ float red[128];
  int b = blockIdx.x, tid = threadIdx.x;
  float m = -3.402823466e38f;
  for (int t = tid; t < T; t += blockDim.x) m = fmaxf(m, s[(size_t)b * T + t]);
  red[tid] = m; __syncthreads();
  for (int k = 64; k > 0; k >>= 1) { if (tid < k) red[tid] = fmaxf(red[tid], red[tid + k]); __syncthreads(); }
  m = red[0]; __syncthreads();
  float sum = 0.f;
  for (int t = tid; t < T; t += blockDim.x) {
    float e = __expf(s[(size_t)b * T + t] - m);
    s[(size_t)b * T + t] = e; sum += e;
  }
  red[tid] = sum; __syncthreads();
  for (int k = 64; k > 0; k >>= 1) { if (tid < k) red[tid] += red[tid + k]; __syncthreads(); }
  if (tid == 0) { smax[b] = m; esum[b] = red[0]; }
}

__global__ __launch_bounds__(256) void exp_inplace_kernel(float* __restrict__ s, int n) {
  int i = blockIdx.x * blockDim.x + threadIdx.x;
  if (i < n) s[i] = __expf(s[i]);
}

// =====================================================================
// Kernel 5: scatter-add  scat[b, idx[t,b]] += e[b,t]  (idx != 0)
// =====================================================================
__global__ __launch_bounds__(256) void scatter_kernel(
    const float* __restrict__ e,   // (B,T)
    const int* __restrict__ idx,   // (T,B)
    float* __restrict__ scat,      // (B,V)
    int T) {
  int i = blockIdx.x * blockDim.x + threadIdx.x;
  if (i >= T * B_DIM) return;
  int t = i / B_DIM, b = i % B_DIM;
  int v = idx[(size_t)t * B_DIM + b];
  if (v != 0) atomicAdd(&scat[(size_t)b * V_DIM + v], e[(size_t)b * T + t]);
}

// =====================================================================
// Kernel 6: pv dot — pvdot[b,v] = EPS*s3e[b,0] + sum_{t>=1} s3e[b,t]*pz[t-1,b,v]
// This streams the 524 MB pv_pz_proba tensor once (the memory roofline term).
// =====================================================================
__global__ __launch_bounds__(256) void pv_dot_kernel(
    const float* __restrict__ s3e,  // (B,TZ)
    const float* __restrict__ pz,   // (TZ,B,V)
    float* __restrict__ pvdot) {    // (B,V)
  int i = blockIdx.x * blockDim.x + threadIdx.x;
  if (i >= B_DIM * V_DIM) return;
  int b = i / V_DIM, v = i % V_DIM;
  float acc = EPSF * s3e[(size_t)b * TZ_DIM];
  for (int t = 1; t < TZ_DIM; ++t) {
    const float* p = pz + (size_t)(t - 1) * B_DIM * V_DIM + (size_t)b * V_DIM + v;
    __builtin_prefetch(p + (size_t)B_DIM * V_DIM, 0, 0);  // next-t line
    acc = fmaf(s3e[(size_t)b * TZ_DIM + t], *p, acc);
  }
  pvdot[i] = acc;
}

// =====================================================================
// Kernel 7: fused softmax over concat [gen | u_copy | m_copy | pv_copy] (B,4V)
// proba[b,v] = sum of the 4 softmax segments at column v. One block per b.
// =====================================================================
__device__ __forceinline__ float4 four_scores(
    size_t o, float gen, const float* us, const float* ms, const float* pv,
    float su, float eu, float sm2, float em) {
  float4 r;
  r.x = gen;
  r.y = __logf(us[o] * (1.f - EPSF) + EPSF * eu) + su;
  r.z = __logf(ms[o] * (1.f - EPSF) + EPSF * em) + sm2;
  r.w = __logf(pv[o]);
  return r;
}

__global__ __launch_bounds__(256) void softmax_proba_kernel(
    const float* __restrict__ gen, const float* __restrict__ uscat,
    const float* __restrict__ mscat, const float* __restrict__ pvdot,
    const float* __restrict__ smax_u, const float* __restrict__ esum_u,
    const float* __restrict__ smax_m, const float* __restrict__ esum_m,
    float* __restrict__ proba, unsigned short* __restrict__ proba_bf) {
  __shared__ float red[256];
  int b = blockIdx.x, tid = threadIdx.x;
  float su = smax_u[b], eu = esum_u[b], sm2 = smax_m[b], em = esum_m[b];
  size_t base = (size_t)b * V_DIM;

  float mx = -3.402823466e38f;
  for (int v = tid; v < V_DIM; v += blockDim.x) {
    float4 s = four_scores(base + v, gen[base + v], uscat, mscat, pvdot, su, eu, sm2, em);
    mx = fmaxf(mx, fmaxf(fmaxf(s.x, s.y), fmaxf(s.z, s.w)));
  }
  red[tid] = mx; __syncthreads();
  for (int k = 128; k > 0; k >>= 1) { if (tid < k) red[tid] = fmaxf(red[tid], red[tid + k]); __syncthreads(); }
  mx = red[0]; __syncthreads();

  float sum = 0.f;
  for (int v = tid; v < V_DIM; v += blockDim.x) {
    float4 s = four_scores(base + v, gen[base + v], uscat, mscat, pvdot, su, eu, sm2, em);
    sum += __expf(s.x - mx) + __expf(s.y - mx) + __expf(s.z - mx) + __expf(s.w - mx);
  }
  red[tid] = sum; __syncthreads();
  for (int k = 128; k > 0; k >>= 1) { if (tid < k) red[tid] += red[tid + k]; __syncthreads(); }
  float inv = 1.f / red[0];

  for (int v = tid; v < V_DIM; v += blockDim.x) {
    float4 s = four_scores(base + v, gen[base + v], uscat, mscat, pvdot, su, eu, sm2, em);
    float p = (__expf(s.x - mx) + __expf(s.y - mx) + __expf(s.z - mx) + __expf(s.w - mx)) * inv;
    proba[base + v] = p;
    proba_bf[base + v] = f2bfu(p);
  }
}

// =====================================================================
// Kernel 8: appr = proba @ mu^T + mu_b ; lsig = proba @ ls^T + ls_b
// M=64, N=512, K=32000 — WMMA bf16, K-loop of 1000 steps. 8 blocks.
// =====================================================================
__global__ __launch_bounds__(256) void final_matmul_kernel(
    const unsigned short* __restrict__ proba_bf,  // (B,V) bf16
    const float* __restrict__ mu_W, const float* __restrict__ mu_b,
    const float* __restrict__ ls_W, const float* __restrict__ ls_b,
    float* __restrict__ appr, float* __restrict__ lsig) {  // (B,E) each
  int lane = threadIdx.x & 31, wid = threadIdx.x >> 5;
  int mat = blockIdx.x >> 2;             // 0 = mu, 1 = ls
  const float* W    = mat ? ls_W : mu_W;
  const float* bias = mat ? ls_b : mu_b;
  float* out        = mat ? lsig : appr;
  int n0 = ((blockIdx.x & 3) * 8 + wid) * 16;
  int n  = n0 + (lane & 15);
  v8f acc[4] = {};
  for (int kk = 0; kk < V_DIM; kk += 32) {
    const float* brow = W + (size_t)n * V_DIM + kk + ((lane & 16) ? 8 : 0);
    __builtin_prefetch(brow + 32, 0, 1);
    v16bf bf = load_b_f32(brow);
    int kb = kk + ((lane & 16) ? 8 : 0);
#pragma unroll
    for (int mt = 0; mt < 4; ++mt) {
      v16bf a = load_a_bf16(proba_bf, V_DIM, mt * 16 + (lane & 15), kb);
      acc[mt] = __builtin_amdgcn_wmma_f32_16x16x32_bf16(
          false, a, false, bf, (short)0, acc[mt], false, false);
    }
  }
  float bv = bias[n];
#pragma unroll
  for (int mt = 0; mt < 4; ++mt)
#pragma unroll
    for (int r = 0; r < 8; ++r) {
      int row = mt * 16 + r + ((lane & 16) ? 8 : 0);
      out[(size_t)row * E_DIM + n] = acc[mt][r] + bv;
    }
}

// =====================================================================
// Kernel 9: sampled = appr + exp(lsig) * eps
// =====================================================================
__global__ __launch_bounds__(256) void sampled_kernel(
    const float* __restrict__ appr, const float* __restrict__ lsig,
    const float* __restrict__ eps, float* __restrict__ sampled) {
  int i = blockIdx.x * blockDim.x + threadIdx.x;
  if (i < B_DIM * E_DIM) sampled[i] = fmaf(__expf(lsig[i]), eps[i], appr[i]);
}

// =====================================================================
// Launch
// =====================================================================
extern "C" void kernel_launch(void* const* d_in, const int* in_sizes, int n_in,
                              void* d_out, int out_size, void* d_ws, size_t ws_size,
                              hipStream_t stream) {
  (void)in_sizes; (void)n_in; (void)out_size; (void)ws_size;
  const float* u_enc   = (const float*)d_in[1];
  const float* pz      = (const float*)d_in[2];
  const float* z_dec   = (const float*)d_in[3];
  const float* m_enc   = (const float*)d_in[5];
  const float* embed_z = (const float*)d_in[6];
  const float* last_h  = (const float*)d_in[7];
  const float* r_eps   = (const float*)d_in[8];
  const int*   u_idx   = (const int*)d_in[9];
  const int*   m_idx   = (const int*)d_in[10];
  const float* W_ih = (const float*)d_in[11];
  const float* W_hh = (const float*)d_in[12];
  const float* b_ih = (const float*)d_in[13];
  const float* b_hh = (const float*)d_in[14];
  const float* w1_W = (const float*)d_in[15];
  const float* w1_b = (const float*)d_in[16];
  const float* mu_W = (const float*)d_in[17];
  const float* mu_b = (const float*)d_in[18];
  const float* ls_W = (const float*)d_in[19];
  const float* ls_b = (const float*)d_in[20];
  const float* p1_W = (const float*)d_in[21];
  const float* p1_b = (const float*)d_in[22];
  const float* p2_W = (const float*)d_in[23];
  const float* p2_b = (const float*)d_in[24];
  const float* p3_W = (const float*)d_in[25];
  const float* p3_b = (const float*)d_in[26];

  // output layout (concatenated return order)
  float* out        = (float*)d_out;
  float* sampled    = out;                               // B*E
  float* gru_out    = sampled + B_DIM * E_DIM;           // B*H
  float* new_hidden = gru_out + B_DIM * H_DIM;           // B*H
  float* proba      = new_hidden + B_DIM * H_DIM;        // B*V
  float* appr       = proba + (size_t)B_DIM * V_DIM;     // B*E
  float* lsig       = appr + B_DIM * E_DIM;              // B*E

  // workspace carve-up
  char* w = (char*)d_ws;
  auto carve = [&](size_t bytes) -> void* {
    void* p = (void*)w; w += (bytes + 255) & ~(size_t)255; return p;
  };
  float*          g        = (float*)carve((size_t)B_DIM * H_DIM * 4);
  unsigned short* g_bf     = (unsigned short*)carve((size_t)B_DIM * H_DIM * 2);
  float*          gen      = (float*)carve((size_t)B_DIM * V_DIM * 4);
  float*          s_u      = (float*)carve((size_t)B_DIM * TU_DIM * 4);
  float*          s_m      = (float*)carve((size_t)B_DIM * TM_DIM * 4);
  float*          s_z      = (float*)carve((size_t)B_DIM * TZ_DIM * 4);
  float*          smax_u   = (float*)carve(B_DIM * 4);
  float*          esum_u   = (float*)carve(B_DIM * 4);
  float*          smax_m   = (float*)carve(B_DIM * 4);
  float*          esum_m   = (float*)carve(B_DIM * 4);
  float*          u_scat   = (float*)carve((size_t)B_DIM * V_DIM * 4);
  float*          m_scat   = (float*)carve((size_t)B_DIM * V_DIM * 4);
  float*          pvdot    = (float*)carve((size_t)B_DIM * V_DIM * 4);
  unsigned short* proba_bf = (unsigned short*)carve((size_t)B_DIM * V_DIM * 2);

  gru_kernel<<<(B_DIM * H_DIM) / 256, 256, 0, stream>>>(
      embed_z, last_h, W_ih, W_hh, b_ih, b_hh, g, g_bf, gru_out, new_hidden);

  gen_score_kernel<<<V_DIM / 128, 256, 0, stream>>>(g_bf, w1_W, w1_b, gen);

  copy_score_kernel<<<(TU_DIM * B_DIM) / 16, 256, 0, stream>>>(u_enc, p1_W, p1_b, g, s_u, TU_DIM);
  copy_score_kernel<<<(TM_DIM * B_DIM) / 16, 256, 0, stream>>>(m_enc, p2_W, p2_b, g, s_m, TM_DIM);
  copy_score_kernel<<<(TZ_DIM * B_DIM) / 16, 256, 0, stream>>>(z_dec, p3_W, p3_b, g, s_z, TZ_DIM);

  softexp_kernel<<<B_DIM, 128, 0, stream>>>(s_u, smax_u, esum_u, TU_DIM);
  softexp_kernel<<<B_DIM, 128, 0, stream>>>(s_m, smax_m, esum_m, TM_DIM);
  exp_inplace_kernel<<<(B_DIM * TZ_DIM + 255) / 256, 256, 0, stream>>>(s_z, B_DIM * TZ_DIM);

  (void)hipMemsetAsync(u_scat, 0, (size_t)B_DIM * V_DIM * 4, stream);
  (void)hipMemsetAsync(m_scat, 0, (size_t)B_DIM * V_DIM * 4, stream);
  scatter_kernel<<<(TU_DIM * B_DIM + 255) / 256, 256, 0, stream>>>(s_u, u_idx, u_scat, TU_DIM);
  scatter_kernel<<<(TM_DIM * B_DIM + 255) / 256, 256, 0, stream>>>(s_m, m_idx, m_scat, TM_DIM);

  pv_dot_kernel<<<((size_t)B_DIM * V_DIM + 255) / 256, 256, 0, stream>>>(s_z, pz, pvdot);

  softmax_proba_kernel<<<B_DIM, 256, 0, stream>>>(
      gen, u_scat, m_scat, pvdot, smax_u, esum_u, smax_m, esum_m, proba, proba_bf);

  final_matmul_kernel<<<8, 256, 0, stream>>>(proba_bf, mu_W, mu_b, ls_W, ls_b, appr, lsig);

  sampled_kernel<<<(B_DIM * E_DIM + 255) / 256, 256, 0, stream>>>(appr, lsig, r_eps, sampled);
}
